// appnp_66907000537298
// MI455X (gfx1250) — compile-verified, matrix operand order checked by reading, not run on hardware
//
#include <hip/hip_runtime.h>

typedef __attribute__((ext_vector_type(16))) _Float16 v16h;
typedef __attribute__((ext_vector_type(8)))  float    v8f;

#define ALPHA_F 0.1f
#define BETA_F  0.9f   // 1 - alpha

// ---------------------------------------------------------------------------
// Repack W [256x64] f32 -> f16 WMMA B-fragments, fragment-ordered so each
// lane reads its 16 halves (32B) contiguously in the GEMM loop.
// ---------------------------------------------------------------------------
__global__ void prep_w_kernel(const float* __restrict__ W, _Float16* __restrict__ wfrag) {
  int flat = blockIdx.x * blockDim.x + threadIdx.x;   // [0, 4 tiles * 8 kchunks * 32 lanes)
  if (flat >= 4 * 8 * 32) return;
  int lane = flat & 31;
  int kc   = (flat >> 5) & 7;
  int t    = flat >> 8;
  int g    = lane >> 4;
  int col  = t * 16 + (lane & 15);
  int k0   = kc * 32;
#pragma unroll
  for (int j = 0; j < 16; ++j) {
    int k = k0 + ((j < 8) ? (g * 8 + j) : (16 + g * 8 + (j - 8)));
    wfrag[flat * 16 + j] = (_Float16)W[k * 64 + col];
  }
}

// ---------------------------------------------------------------------------
// h0 = relu(x @ W + b) via v_wmma_f32_16x16x32_f16.
// Block = 128 threads = 4 waves; wave t computes the 16x16 tile (row0, 16t).
// ---------------------------------------------------------------------------
__global__ void gemm_relu_wmma(const float* __restrict__ x,
                               const _Float16* __restrict__ wfrag,
                               const float* __restrict__ bias,
                               float* __restrict__ h0, int Nnodes) {
  int wave = threadIdx.x >> 5;        // output col-tile 0..3
  int lane = threadIdx.x & 31;
  int g    = lane >> 4;
  int r    = lane & 15;
  int row0 = blockIdx.x * 16;
  int row  = row0 + r;
  int rowc = row < Nnodes ? row : Nnodes - 1;   // clamp (stores are guarded)

  v8f c = {};
  const _Float16* wp = wfrag + (size_t)wave * 8 * 32 * 16;
  for (int kc = 0; kc < 8; ++kc) {
    const float* xp = x + (size_t)rowc * 256 + kc * 32 + g * 8;
    v16h a;
#pragma unroll
    for (int j = 0; j < 8; ++j) a[j] = (_Float16)xp[j];
#pragma unroll
    for (int j = 0; j < 8; ++j) a[8 + j] = (_Float16)xp[16 + j];
    v16h bmat = *(const v16h*)(wp + (kc * 32 + lane) * 16);
    c = __builtin_amdgcn_wmma_f32_16x16x32_f16(false, a, false, bmat,
                                               (short)0, c, false, false);
  }
  int col = wave * 16 + r;
  float bb = bias[col];
#pragma unroll
  for (int v = 0; v < 8; ++v) {
    int orow = row0 + v + 8 * g;     // C layout: VGPR v -> M = v + 8*g
    if (orow < Nnodes) {
      float val = c[v] + bb;
      h0[(size_t)orow * 64 + col] = val > 0.f ? val : 0.f;
    }
  }
}

// --------------------------- CSR construction ------------------------------
__global__ void zero_u32_kernel(unsigned* __restrict__ p, int n) {
  int i = blockIdx.x * blockDim.x + threadIdx.x;
  if (i < n) p[i] = 0u;
}

__global__ void cnt_kernel(const int* __restrict__ cols, unsigned* __restrict__ cnt, int E) {
  int e = blockIdx.x * blockDim.x + threadIdx.x;
  if (e < E) atomicAdd(&cnt[cols[e]], 1u);       // global_atomic_add_u32 (one-time)
}

__global__ void dinv_kernel(const unsigned* __restrict__ cnt, float* __restrict__ dinv, int N) {
  int i = blockIdx.x * blockDim.x + threadIdx.x;
  if (i < N) dinv[i] = rsqrtf((float)(cnt[i] + 1u));   // +1 self-loop
}

// Block-local exclusive scan (256) + per-block totals.
__global__ void scan1_kernel(const unsigned* __restrict__ cnt, unsigned* __restrict__ part,
                             unsigned* __restrict__ bsum, int N) {
  __shared__ unsigned s[256];
  int t = threadIdx.x;
  int i = blockIdx.x * 256 + t;
  unsigned v = (i < N) ? cnt[i] : 0u;
  s[t] = v;
  __syncthreads();
  for (int off = 1; off < 256; off <<= 1) {
    unsigned u = (t >= off) ? s[t - off] : 0u;
    __syncthreads();
    s[t] += u;
    __syncthreads();
  }
  if (i < N) part[i] = s[t] - v;                 // exclusive within block
  if (t == 255) bsum[blockIdx.x] = s[255];       // block total
}

// Single-block exclusive scan of block totals (nb <= 1024).
__global__ void scan2_kernel(unsigned* __restrict__ bsum, int nb) {
  __shared__ unsigned s[1024];
  int t = threadIdx.x;
  unsigned v = (t < nb) ? bsum[t] : 0u;
  s[t] = v;
  __syncthreads();
  for (int off = 1; off < 1024; off <<= 1) {
    unsigned u = (t >= off) ? s[t - off] : 0u;
    __syncthreads();
    s[t] += u;
    __syncthreads();
  }
  if (t < nb) bsum[t] = s[t] - v;                // exclusive, in place
}

__global__ void scan3_kernel(const unsigned* __restrict__ part, const unsigned* __restrict__ bsum,
                             unsigned* __restrict__ rowptr, int N, int E) {
  int i = blockIdx.x * blockDim.x + threadIdx.x;
  if (i < N) rowptr[i] = part[i] + bsum[i >> 8];
  if (i == 0) rowptr[N] = (unsigned)E;
}

// Scatter edges into CSR slots; fold (1-alpha)*dinv[src]*dinv[dst] into weight.
__global__ void fill_kernel(const int* __restrict__ rows, const int* __restrict__ cols,
                            const float* __restrict__ dinv, const unsigned* __restrict__ rowptr,
                            unsigned* __restrict__ fill, int* __restrict__ src_sorted,
                            float* __restrict__ wnorm, int E) {
  int e = blockIdx.x * blockDim.x + threadIdx.x;
  if (e >= E) return;
  int r = rows[e];
  int c = cols[e];
  unsigned pos = rowptr[c] + atomicAdd(&fill[c], 1u);
  src_sorted[pos] = r;
  wnorm[pos] = BETA_F * dinv[r] * dinv[c];
}

// --------------------------- propagation (pull) ----------------------------
// One wave per node. Lane owns channels (2*lane, 2*lane+1).
// h_next[i] = alpha*h0[i] + (1-a)*dinv[i]^2*h_cur[i] + sum_e w[e]*h_cur[src[e]]
// Edge metadata loaded coalesced (32 edges at a time) and broadcast by shfl.
__global__ void gather_step_kernel(const unsigned* __restrict__ rowptr,
                                   const int* __restrict__ src_sorted,
                                   const float* __restrict__ wnorm,
                                   const float* __restrict__ h0,
                                   const float* __restrict__ h_cur,
                                   const float* __restrict__ dinv,
                                   float* __restrict__ h_next, int N) {
  int wave = threadIdx.x >> 5;
  int lane = threadIdx.x & 31;
  int node = blockIdx.x * 8 + wave;
  if (node >= N) return;

  unsigned start = rowptr[node];
  unsigned end   = rowptr[node + 1];

  size_t cbase = (size_t)node * 64 + lane * 2;
  float2 hc  = *(const float2*)(h_cur + cbase);
  float2 hz  = *(const float2*)(h0 + cbase);
  float  d   = dinv[node];
  float  sw  = BETA_F * d * d;                    // self-loop weight
  float  ax  = ALPHA_F * hz.x + sw * hc.x;
  float  ay  = ALPHA_F * hz.y + sw * hc.y;

  for (unsigned base = start; base < end; base += 32) {
    unsigned idx = base + lane;
    int   s = 0;
    float w = 0.f;
    if (idx < end) {                              // coalesced metadata load
      s = src_sorted[idx];
      w = wnorm[idx];
    }
    int m = (int)(end - base);
    if (m > 32) m = 32;
    for (int i = 0; i < m; ++i) {
      int   ss = __shfl(s, i, 32);
      float ww = __shfl(w, i, 32);
      float2 hv = *(const float2*)(h_cur + (size_t)ss * 64 + lane * 2);
      ax += ww * hv.x;
      ay += ww * hv.y;
    }
  }
  float2 o;
  o.x = ax;
  o.y = ay;
  *(float2*)(h_next + cbase) = o;
}

// --------------------------- log-softmax -----------------------------------
__global__ void logsoftmax_kernel(const float* __restrict__ h, float* __restrict__ out, int N) {
  int wave = threadIdx.x >> 5;
  int lane = threadIdx.x & 31;
  int node = blockIdx.x * 8 + wave;
  if (node >= N) return;
  float2 v = *(const float2*)(h + (size_t)node * 64 + lane * 2);
  float m = fmaxf(v.x, v.y);
#pragma unroll
  for (int s = 16; s >= 1; s >>= 1) m = fmaxf(m, __shfl_xor(m, s, 32));
  float ex = __expf(v.x - m) + __expf(v.y - m);
#pragma unroll
  for (int s = 16; s >= 1; s >>= 1) ex += __shfl_xor(ex, s, 32);
  float lse = m + __logf(ex);
  out[(size_t)node * 64 + lane * 2]     = v.x - lse;
  out[(size_t)node * 64 + lane * 2 + 1] = v.y - lse;
}

// ---------------------------------------------------------------------------
extern "C" void kernel_launch(void* const* d_in, const int* in_sizes, int n_in,
                              void* d_out, int out_size, void* d_ws, size_t ws_size,
                              hipStream_t stream) {
  const float* x    = (const float*)d_in[0];
  const int*   ei   = (const int*)d_in[1];
  const float* W    = (const float*)d_in[2];
  const float* bias = (const float*)d_in[3];

  const int N = in_sizes[0] / 256;
  const int E = in_sizes[1] / 2;
  const int* rows = ei;        // edge_index[0] = sources
  const int* cols = ei + E;    // edge_index[1] = targets

  // ---- workspace carve-up (~105 MB) ----
  char* ws = (char*)d_ws;
  size_t off = 0;
  auto take = [&](size_t bytes) -> char* {
    off = (off + 255) & ~(size_t)255;
    char* p = ws + off;
    off += bytes;
    return p;
  };
  float*     h0     = (float*)take((size_t)N * 64 * sizeof(float));
  float*     hA     = (float*)take((size_t)N * 64 * sizeof(float));
  float*     hB     = (float*)take((size_t)N * 64 * sizeof(float));
  unsigned*  cnt    = (unsigned*)take((size_t)N * sizeof(unsigned));
  unsigned*  fill   = (unsigned*)take((size_t)N * sizeof(unsigned));
  float*     dinv   = (float*)take((size_t)N * sizeof(float));
  unsigned*  part   = (unsigned*)take((size_t)N * sizeof(unsigned));
  unsigned*  bsum   = (unsigned*)take((size_t)1024 * sizeof(unsigned));
  unsigned*  rowptr = (unsigned*)take((size_t)(N + 1) * sizeof(unsigned));
  int*       srcs   = (int*)take((size_t)E * sizeof(int));
  float*     wnorm  = (float*)take((size_t)E * sizeof(float));
  _Float16*  wfrag  = (_Float16*)take((size_t)4 * 8 * 32 * 16 * sizeof(_Float16));
  (void)ws_size;

  const int T  = 256;
  const int nb = (N + 255) / 256;    // #scan blocks (391 for N=100000, must be <=1024)

  // ---- one-time precompute (amortized over 50 steps) ----
  prep_w_kernel<<<(4 * 8 * 32 + T - 1) / T, T, 0, stream>>>(W, wfrag);
  zero_u32_kernel<<<(N + T - 1) / T, T, 0, stream>>>(cnt, N);
  zero_u32_kernel<<<(N + T - 1) / T, T, 0, stream>>>(fill, N);
  cnt_kernel<<<(E + T - 1) / T, T, 0, stream>>>(cols, cnt, E);
  dinv_kernel<<<(N + T - 1) / T, T, 0, stream>>>(cnt, dinv, N);
  scan1_kernel<<<nb, 256, 0, stream>>>(cnt, part, bsum, N);
  scan2_kernel<<<1, 1024, 0, stream>>>(bsum, nb);
  scan3_kernel<<<(N + T - 1) / T, T, 0, stream>>>(part, bsum, rowptr, N, E);
  fill_kernel<<<(E + T - 1) / T, T, 0, stream>>>(rows, cols, dinv, rowptr, fill, srcs, wnorm, E);
  gemm_relu_wmma<<<(N + 15) / 16, 128, 0, stream>>>(x, wfrag, bias, h0, N);

  // ---- K = 50 propagation steps, ping-pong, atomic-free pull ----
  float* cur = h0;
  float* nxt = hA;
  for (int step = 0; step < 50; ++step) {
    gather_step_kernel<<<(N + 7) / 8, T, 0, stream>>>(rowptr, srcs, wnorm, h0, cur, dinv, nxt, N);
    cur = nxt;
    nxt = (cur == hA) ? hB : hA;
  }

  logsoftmax_kernel<<<(N + 7) / 8, T, 0, stream>>>(cur, (float*)d_out, N);
}
